// DecoderConv_36318243455331
// MI455X (gfx1250) — matrix operand order repaired and checked by reading, not compiled, not run on hardware
//
#include <hip/hip_runtime.h>
#include <stdint.h>

// Problem constants (from reference): B=4, Cin=Cout=64, L=131072=2^17, K=9
#define B_    4
#define C_    64
#define L_    131072
#define LLOG_ 17
#define K_    9
#define PAD_  4
#define KC_   (C_ * K_)            // 576 = GEMM K
#define TILE_ 64                   // L-columns per block
#define XW_   (TILE_ + 2 * PAD_)   // 72 (tile + halo)
#define BS_STRIDE 584              // f16 row stride for B panel (16B aligned rows)
#define NBLK_ (B_ * (L_ / TILE_))  // 8192 blocks

typedef _Float16 h8   __attribute__((ext_vector_type(8)));
typedef _Float16 v16h __attribute__((ext_vector_type(16)));
typedef float    v8f  __attribute__((ext_vector_type(8)));
typedef uint32_t u32x4 __attribute__((ext_vector_type(4)));
typedef uint32_t u32x8 __attribute__((ext_vector_type(8)));

static __device__ __forceinline__ v16h cat16(h8 lo, h8 hi) {
  return __builtin_shufflevector(lo, hi, 0,1,2,3,4,5,6,7,8,9,10,11,12,13,14,15);
}

// ---- kernel 0: W (f32 [64][576]) -> f16 copy in workspace ------------------
__global__ void wconv_prepW(const float* __restrict__ W, _Float16* __restrict__ Wh) {
  int i = blockIdx.x * blockDim.x + threadIdx.x;
  if (i < C_ * KC_) Wh[i] = (_Float16)W[i];
}

// ---- kernel 1: weighted conv via WMMA + per-block stats --------------------
__global__ __launch_bounds__(128) void wconv_main(
    const float* __restrict__ x, const float* __restrict__ coords,
    const _Float16* __restrict__ Wh, const float* __restrict__ bias,
    float* __restrict__ out, float* __restrict__ psum, float* __restrict__ psq)
{
  __shared__ float    xs[C_ * XW_];          // 18,432 B  x tile (TDM dest, dense rows)
  __shared__ float    cs[3 * XW_];           //    864 B  coord tile (TDM dest)
  __shared__ float    gws[K_ * TILE_];       //  2,304 B  gaussian weights [k][n]
  __shared__ _Float16 Bs[TILE_ * BS_STRIDE]; // 74,752 B  B panel, [n][kidx]
  __shared__ float    outs[C_ * (TILE_+1)];  // 16,640 B  out tile for stats

  const int tid = threadIdx.x;
  const int b   = blockIdx.x / (L_ / TILE_);
  const int l0  = (blockIdx.x % (L_ / TILE_)) * TILE_;

  const bool interior = (l0 >= PAD_) && (l0 + TILE_ + PAD_ <= L_);
  if (interior) {
    // --- Tensor Data Mover: 2D tiles [rows][XW_] from global (row stride L) to LDS
    if (tid < 32) {                      // wave 0 issues the DMAs
      uint64_t gx = (uint64_t)(uintptr_t)(x + (((uint64_t)b * C_) << LLOG_) + (l0 - PAD_));
      u32x4 g0; u32x8 g1;
      g0[0] = 1u;                                        // count=1, user descriptor
      g0[1] = (uint32_t)(uintptr_t)&xs[0];               // lds_addr
      g0[2] = (uint32_t)gx;                              // global_addr[31:0]
      g0[3] = ((uint32_t)(gx >> 32) & 0x01FFFFFFu) | (2u << 30); // addr[56:32] | type=2
      g1[0] = 0x00020000u;                               // data_size=4B, no pad/iter/mask
      g1[1] = ((uint32_t)L_ & 0xFFFFu) << 16;            // tensor_dim0[15:0]
      g1[2] = ((uint32_t)L_ >> 16) | ((uint32_t)C_ << 16); // dim0[31:16] | dim1[15:0]
      g1[3] = ((uint32_t)XW_ << 16);                     // dim1[31:16]=0 | tile_dim0
      g1[4] = (uint32_t)C_;                              // tile_dim1 | tile_dim2=0
      g1[5] = (uint32_t)L_;                              // tensor_dim0_stride[31:0]
      g1[6] = 0u; g1[7] = 0u;                            // stride hi / dim1_stride (2D)
      asm volatile("tensor_load_to_lds %0, %1" :: "s"(g0), "s"(g1) : "memory");

      uint64_t gc = (uint64_t)(uintptr_t)(coords + (((uint64_t)b * 3) << LLOG_) + (l0 - PAD_));
      u32x4 c0 = g0; u32x8 c1 = g1;
      c0[1] = (uint32_t)(uintptr_t)&cs[0];
      c0[2] = (uint32_t)gc;
      c0[3] = ((uint32_t)(gc >> 32) & 0x01FFFFFFu) | (2u << 30);
      c1[2] = ((uint32_t)L_ >> 16) | (3u << 16);         // tensor_dim1 = 3
      c1[4] = 3u;                                        // tile_dim1  = 3
      asm volatile("tensor_load_to_lds %0, %1" :: "s"(c0), "s"(c1) : "memory");
      __builtin_amdgcn_s_wait_tensorcnt(0);
    }
  } else {
    // --- edge tiles: guarded cooperative loads with zero padding
    for (int i = tid; i < C_ * XW_; i += 128) {
      int ci = i / XW_, col = i - ci * XW_;
      int l = l0 - PAD_ + col;
      xs[i] = (l >= 0 && l < L_) ? x[(((uint64_t)(b * C_ + ci)) << LLOG_) + l] : 0.f;
    }
    for (int i = tid; i < 3 * XW_; i += 128) {
      int d = i / XW_, col = i - d * XW_;
      int l = l0 - PAD_ + col;
      cs[i] = (l >= 0 && l < L_) ? coords[(((uint64_t)(b * 3 + d)) << LLOG_) + l] : 0.f;
    }
  }
  __syncthreads();

  // --- gaussian spatial weights: gws[k][n] = exp(-0.5 * ||c(n+k-4)-c(n)||^2)
  for (int i = tid; i < K_ * TILE_; i += 128) {
    int n = i & (TILE_ - 1), k = i >> 6;
    float d2 = 0.f;
#pragma unroll
    for (int d = 0; d < 3; d++) {
      float diff = cs[d * XW_ + n + k] - cs[d * XW_ + n + PAD_];
      d2 += diff * diff;
    }
    gws[i] = __expf(-0.5f * d2);
  }
  __syncthreads();

  // --- build f16 B panel: Bs[n][kidx] = x[ci][n+k] * gws[k][n], kidx = ci*9+k
  for (int i = tid; i < KC_ * TILE_; i += 128) {
    int n = i & (TILE_ - 1), kidx = i >> 6;
    int ci = kidx / K_, k = kidx - ci * K_;
    Bs[n * BS_STRIDE + kidx] = (_Float16)(xs[ci * XW_ + n + k] * gws[k * TILE_ + n]);
  }
  __syncthreads();

  // --- WMMA: each wave does a 64(M) x 16(N) tile, K = 576 in 18 steps of 32
  const int lane = tid & 31;
  const int wsl  = tid >> 5;                 // wave's 16-column slice
  const int m    = lane & 15;
  const int kbA  = (lane < 16) ? 0 : 16;     // A: K 0..15 vs 16..31
  const int kbB  = (lane < 16) ? 0 : 16;     // B: K 0..15 vs 16..31
  const int ncol = (wsl << 4) + m;
  v8f acc[4] = {};
  for (int kk = 0; kk < KC_; kk += 32) {
    const h8* bp = (const h8*)&Bs[ncol * BS_STRIDE + kk + kbB];
    v16h bf = cat16(bp[0], bp[1]);
#pragma unroll
    for (int mt = 0; mt < 4; mt++) {
      const _Float16* wp = Wh + (size_t)((mt << 4) + m) * KC_ + kk + kbA;
      v16h af = cat16(*(const h8*)wp, *(const h8*)(wp + 8));
      acc[mt] = __builtin_amdgcn_wmma_f32_16x16x32_f16(
          false, af, false, bf, (short)0, acc[mt], false, false);
    }
  }

  // --- write pre-BN output + stage tile for deterministic stats
  const int hb = (lane >> 4) << 3;           // C/D layout: lanes 16-31 hold M = r+8
#pragma unroll
  for (int mt = 0; mt < 4; mt++) {
#pragma unroll
    for (int r = 0; r < 8; r++) {
      int co = (mt << 4) + hb + r;
      float v = acc[mt][r] + bias[co];
      out[(((uint64_t)(b * C_ + co)) << LLOG_) + l0 + ncol] = v;
      outs[co * (TILE_ + 1) + ncol] = v;
    }
  }
  __syncthreads();
  if (tid < C_) {                            // fixed-order per-channel reduction
    float s = 0.f, q = 0.f;
    for (int n = 0; n < TILE_; n++) {
      float v = outs[tid * (TILE_ + 1) + n];
      s += v; q += v * v;
    }
    psum[(uint64_t)blockIdx.x * C_ + tid] = s;
    psq [(uint64_t)blockIdx.x * C_ + tid] = q;
  }
}

// ---- kernel 2: fold per-block partials -> per-channel scale/shift ----------
__global__ void wconv_finalize(const float* __restrict__ psum, const float* __restrict__ psq,
                               const float* __restrict__ gamma, const float* __restrict__ beta,
                               float* __restrict__ scale, float* __restrict__ shift) {
  int c = threadIdx.x;
  if (c < C_) {
    float s = 0.f, q = 0.f;
    for (int i = 0; i < NBLK_; i++) { s += psum[(uint64_t)i * C_ + c]; q += psq[(uint64_t)i * C_ + c]; }
    const float N = (float)B_ * (float)L_;
    float mean = s / N;
    float var  = q / N - mean * mean;
    float inv  = rsqrtf(var + 1e-5f);
    float sc   = gamma[c] * inv;
    scale[c] = sc;
    shift[c] = beta[c] - mean * sc;
  }
}

// ---- kernel 3: in-place BN + ReLU (float4) ---------------------------------
__global__ void wconv_bnrelu(float* __restrict__ out,
                             const float* __restrict__ scale, const float* __restrict__ shift) {
  __shared__ float sc[C_], sh[C_];
  if (threadIdx.x < C_) { sc[threadIdx.x] = scale[threadIdx.x]; sh[threadIdx.x] = shift[threadIdx.x]; }
  __syncthreads();
  const size_t total4 = (size_t)B_ * C_ * L_ / 4;
  for (size_t i = (size_t)blockIdx.x * blockDim.x + threadIdx.x; i < total4;
       i += (size_t)gridDim.x * blockDim.x) {
    int ch = (int)((i >> (LLOG_ - 2)) & (C_ - 1));   // L/4 = 2^15 per channel
    float4 v = ((float4*)out)[i];
    float a = sc[ch], c = sh[ch];
    v.x = fmaxf(v.x * a + c, 0.f);
    v.y = fmaxf(v.y * a + c, 0.f);
    v.z = fmaxf(v.z * a + c, 0.f);
    v.w = fmaxf(v.w * a + c, 0.f);
    ((float4*)out)[i] = v;
  }
}

extern "C" void kernel_launch(void* const* d_in, const int* in_sizes, int n_in,
                              void* d_out, int out_size, void* d_ws, size_t ws_size,
                              hipStream_t stream) {
  (void)in_sizes; (void)n_in; (void)out_size; (void)ws_size;
  const float* x      = (const float*)d_in[0];
  const float* coords = (const float*)d_in[1];
  const float* W      = (const float*)d_in[2];
  const float* bias   = (const float*)d_in[3];
  const float* gamma  = (const float*)d_in[4];
  const float* beta   = (const float*)d_in[5];
  float* out = (float*)d_out;

  // workspace layout: Wh f16 [64*576] | scale[64] shift[64] | psum[8192*64] | psq[8192*64]
  char* ws = (char*)d_ws;
  _Float16* Wh = (_Float16*)ws;                       // 73,728 B
  float* scale = (float*)(ws + 73728);                //    256 B
  float* shift = scale + C_;                          //    256 B
  float* psum  = (float*)(ws + 74240);                // 2 MiB
  float* psq   = psum + (size_t)NBLK_ * C_;           // 2 MiB

  wconv_prepW<<<(C_ * KC_ + 255) / 256, 256, 0, stream>>>(W, Wh);
  wconv_main<<<NBLK_, 128, 0, stream>>>(x, coords, Wh, bias, out, psum, psq);
  wconv_finalize<<<1, 64, 0, stream>>>(psum, psq, gamma, beta, scale, shift);
  wconv_bnrelu<<<2048, 256, 0, stream>>>(out, scale, shift);
}
